// DecoderTreeLSTM_25933012533727
// MI455X (gfx1250) — matrix operand order repaired and checked by reading, not compiled, not run on hardware
//
#include <hip/hip_runtime.h>

typedef __attribute__((ext_vector_type(16))) _Float16 v16h;
typedef __attribute__((ext_vector_type(8)))  float    v8f;

#define NOBJ  4096
#define INDIM 1024
#define EMB   200
#define HID   1024
#define NCLS  151
#define INSZ  1224          // INDIM + EMB
#define KTX   39            // ceil(1248/32): K tiles for x-side GEMM (padded K = 1248)
#define KTH   32            // 1024/32: K tiles for h-side GEMM
#define NX    5120          // W_px(1024) | W_ioux(3072) | W_fx(1024) stacked
#define NH    4096          // W_iouh(3072) | W_fh(1024) stacked
#define NTX   (NX / 16)     // 320 output tiles
#define NTH   (NH / 16)     // 256 output tiles
#define CHUNK 256           // max nodes processed per level-chunk
#define MTMAX (CHUNK / 16)
#define FRAG  512           // one 16x32 fragment = 32 lanes * 16 f16
#define GBLK  40
#define TPB   256

// ---------------- workspace layout (bytes) ----------------
static constexpr size_t algn(size_t x) { return (x + 255) & ~(size_t)255; }
static constexpr size_t OFF_CTL = 0;                                                  // ctl[0]=barrier ctl[1]=numChunks
static constexpr size_t OFF_CHK = 256;                                                // chunkStart[NOBJ+2]
static constexpr size_t OFF_ORD = OFF_CHK + algn(sizeof(int) * (NOBJ + 2));           // order[NOBJ]
static constexpr size_t OFF_DEP = OFF_ORD + algn(sizeof(int) * NOBJ);                 // depth[NOBJ]
static constexpr size_t OFF_CNT = OFF_DEP + algn(sizeof(int) * NOBJ);                 // counting-sort scratch
static constexpr size_t OFF_LAB = OFF_CNT + algn(sizeof(int) * (NOBJ + 4));           // labels[NOBJ]
static constexpr size_t OFF_WXF = OFF_LAB + algn(sizeof(int) * NOBJ);                 // f16 Wx fragments
static constexpr size_t OFF_WHF = OFF_WXF + algn((size_t)2 * NTX * KTX * FRAG);       // f16 Wh fragments
static constexpr size_t OFF_HAL = OFF_WHF + algn((size_t)2 * NTH * KTH * FRAG);       // f16 h_all[NOBJ*HID]
static constexpr size_t OFF_CAL = OFF_HAL + algn((size_t)2 * NOBJ * HID);             // f32 c_all[NOBJ*HID]
static constexpr size_t OFF_XF  = OFF_CAL + algn(sizeof(float) * (size_t)NOBJ * HID); // f16 X fragments (chunk)
static constexpr size_t OFF_PF  = OFF_XF + algn((size_t)2 * MTMAX * KTX * FRAG);      // f16 parent-h fragments
static constexpr size_t OFF_Y1  = OFF_PF + algn((size_t)2 * MTMAX * KTH * FRAG);      // f32 Y1[CHUNK*NX]
static constexpr size_t OFF_Y2  = OFF_Y1 + algn(sizeof(float) * (size_t)CHUNK * NX);  // f32 Y2[CHUNK*NH]

// A-fragment K index for (lane, j): 16x32 f16 A layout (ISA 7.12.2)
__device__ __forceinline__ int kofA(int lane, int j) {
    return ((j & 8) << 1) + ((lane & 16) >> 1) + (j & 7);
}
// B-fragment K index for (lane, j): same K order as kofA (lanes 0-15: K=0..7,16..23; 16-31: K=8..15,24..31)
__device__ __forceinline__ int kofB(int lane, int j) { return ((j & 8) << 1) + ((lane & 16) >> 1) + (j & 7); }

__device__ __forceinline__ float sigf(float x) { return 1.0f / (1.0f + __expf(-x)); }

// ---------------- prep: weights -> f16 WMMA B-fragment layout ----------------
__global__ void prep_weights(const float* __restrict__ Wpx, const float* __restrict__ Wioux,
                             const float* __restrict__ Wfx, const float* __restrict__ Wiouh,
                             const float* __restrict__ Wfh,
                             _Float16* __restrict__ WxF, _Float16* __restrict__ WhF) {
    const long totX = (long)NTX * KTX * FRAG;
    const long totH = (long)NTH * KTH * FRAG;
    for (long e = (long)blockIdx.x * blockDim.x + threadIdx.x; e < totX + totH;
         e += (long)gridDim.x * blockDim.x) {
        if (e < totX) {
            long frag = e >> 9; int within = (int)(e & 511);
            int lane = within >> 4, j = within & 15;
            int nt = (int)(frag / KTX), kt = (int)(frag % KTX);
            int n = nt * 16 + (lane & 15);
            int k = kt * 32 + kofB(lane, j);
            float v = 0.0f;
            if (k < INSZ) {
                if (n < HID)              v = Wpx[(long)n * INSZ + k];
                else if (n < 4 * HID)     v = Wioux[(long)(n - HID) * INSZ + k];
                else                      v = Wfx[(long)(n - 4 * HID) * INSZ + k];
            }
            WxF[e] = (_Float16)v;
        } else {
            long e2 = e - totX;
            long frag = e2 >> 9; int within = (int)(e2 & 511);
            int lane = within >> 4, j = within & 15;
            int nt = (int)(frag / KTH), kt = (int)(frag % KTH);
            int n = nt * 16 + (lane & 15);
            int k = kt * 32 + kofB(lane, j);
            float v = (n < 3 * HID) ? Wiouh[(long)n * HID + k]
                                    : Wfh[(long)(n - 3 * HID) * HID + k];
            WhF[e2] = (_Float16)v;
        }
    }
}

// ---------------- prep: depth levels -> chunk schedule (serial, tiny) ----------------
__global__ void prep_levels(const int* __restrict__ parent, int* __restrict__ ctl,
                            int* __restrict__ chk, int* __restrict__ order,
                            int* __restrict__ depth, int* __restrict__ cnt) {
    if (threadIdx.x != 0 || blockIdx.x != 0) return;
    ctl[0] = 0;  // grid barrier counter
    int maxd = 0;
    depth[0] = 0;
    for (int t = 1; t < NOBJ; ++t) {
        int d = depth[parent[t]] + 1;
        depth[t] = d;
        if (d > maxd) maxd = d;
    }
    for (int d = 0; d <= maxd + 1; ++d) cnt[d] = 0;
    for (int t = 0; t < NOBJ; ++t) cnt[depth[t] + 1]++;
    for (int d = 0; d <= maxd; ++d) cnt[d + 1] += cnt[d];
    for (int t = 0; t < NOBJ; ++t) order[cnt[depth[t]]++] = t;  // cnt[d] -> end of level d
    int nch = 0, pos = 0;
    chk[0] = 0;
    for (int d = 0; d <= maxd; ++d) {
        int end = cnt[d];
        while (pos < end) {
            int sz = end - pos;
            if (sz > CHUNK) sz = CHUNK;
            pos += sz;
            chk[++nch] = pos;
        }
    }
    ctl[1] = nch;
}

// ---------------- grid-wide barrier (monotonic counter) ----------------
__device__ __forceinline__ void gsync(unsigned* bar, unsigned target) {
    __syncthreads();
    if (threadIdx.x == 0) {
        __threadfence();
        atomicAdd(bar, 1u);
        while (*(volatile unsigned*)bar < target) __builtin_amdgcn_s_sleep(1);
        __threadfence();
    }
    __syncthreads();
}

// ---------------- persistent level-parallel tree-LSTM ----------------
__global__ void __launch_bounds__(TPB, 1) tree_lstm_main(
    const float* __restrict__ features, const float* __restrict__ embedW,
    const float* __restrict__ b_px, const float* __restrict__ b_ioux,
    const float* __restrict__ b_iouh, const float* __restrict__ b_fx,
    const float* __restrict__ b_fh, const float* __restrict__ Wout,
    const float* __restrict__ b_out, const int* __restrict__ proc_order,
    const int* __restrict__ parent, const int* __restrict__ ctl,
    const int* __restrict__ chk, const int* __restrict__ order,
    int* __restrict__ labels, const _Float16* __restrict__ WxF,
    const _Float16* __restrict__ WhF, _Float16* __restrict__ hAll,
    float* __restrict__ cAll, _Float16* __restrict__ XF, _Float16* __restrict__ PF,
    float* __restrict__ Y1, float* __restrict__ Y2,
    float* __restrict__ outDist, float* __restrict__ outLab, unsigned* __restrict__ bar) {
    const int tid = threadIdx.x;
    const int gtid = blockIdx.x * blockDim.x + tid;
    const int nThreads = gridDim.x * blockDim.x;
    const int lane = tid & 31;
    const int wave = gtid >> 5;
    const int nWaves = nThreads >> 5;
    const int numChunks = ctl[1];
    unsigned target = 0;

    __shared__ float hbuf[HID];
    __shared__ float dbuf[NCLS + 1];

    for (int ch = 0; ch < numChunks; ++ch) {
        const int s0 = chk[ch];
        const int Bc = chk[ch + 1] - s0;
        const int mTiles = (Bc + 15) >> 4;

        // ---- Phase A: gather activations into WMMA A-fragment layout ----
        const long totX = (long)mTiles * KTX * FRAG;
        const long totP = (long)mTiles * KTH * FRAG;
        for (long e = gtid; e < totX + totP; e += nThreads) {
            if (e < totX) {
                long frag = e >> 9; int within = (int)(e & 511);
                int ln = within >> 4, j = within & 15;
                int mt = (int)(frag / KTX), kt = (int)(frag % KTX);
                int m = mt * 16 + (ln & 15);
                int k = kt * 32 + kofA(ln, j);
                float v = 0.0f;
                if (m < Bc) {
                    int t = order[s0 + m];
                    if (k < INDIM) {
                        v = features[(long)proc_order[t] * INDIM + k];
                    } else if (k < INSZ) {
                        int pid = parent[t];
                        int eidx = (pid < 0) ? 0 : (labels[pid] + 1);
                        v = embedW[(long)eidx * EMB + (k - INDIM)];
                    }
                }
                XF[e] = (_Float16)v;
            } else {
                long e2 = e - totX;
                long frag = e2 >> 9; int within = (int)(e2 & 511);
                int ln = within >> 4, j = within & 15;
                int mt = (int)(frag / KTH), kt = (int)(frag % KTH);
                int m = mt * 16 + (ln & 15);
                int k = kt * 32 + kofA(ln, j);
                _Float16 v = (_Float16)0.0f;
                if (m < Bc) {
                    int pid = parent[order[s0 + m]];
                    if (pid >= 0) v = hAll[(long)pid * HID + k];
                }
                PF[e2] = v;
            }
        }
        target += gridDim.x; gsync(bar, target);

        // ---- Phase B: batched GEMM, 2M x 2N register-blocked WMMA ----
        // Each wave owns a 32x32 output tile: 2 A-frags + 2 B-frags -> 4 WMMAs per K-step
        // (2x FLOPs per fragment byte; halves L2 weight re-read traffic per chunk).
        const int mT2 = (mTiles + 1) >> 1;  // 32-row super-tiles (padded pair reads stale
                                            // fragments; its Y rows are never consumed)
        const int jobs1 = mT2 * (NTX / 2);
        const int jobsT = jobs1 + mT2 * (NTH / 2);
        for (int job = wave; job < jobsT; job += nWaves) {
            v8f acc00 = {}, acc01 = {}, acc10 = {}, acc11 = {};
            if (job < jobs1) {
                int ms = job / (NTX / 2), np = job % (NTX / 2);
                const _Float16* a0 = XF + (long)(ms * 2) * KTX * FRAG + lane * 16;
                const _Float16* a1 = XF + (long)(ms * 2 + 1) * KTX * FRAG + lane * 16;
                const _Float16* bq0 = WxF + (long)(np * 2) * KTX * FRAG + lane * 16;
                const _Float16* bq1 = WxF + (long)(np * 2 + 1) * KTX * FRAG + lane * 16;
                for (int kt = 0; kt < KTX; ++kt) {
                    __builtin_prefetch(bq0 + FRAG, 0, 3);
                    __builtin_prefetch(bq1 + FRAG, 0, 3);
                    v16h A0 = *(const v16h*)a0;
                    v16h A1 = *(const v16h*)a1;
                    v16h B0 = *(const v16h*)bq0;
                    v16h B1 = *(const v16h*)bq1;
                    acc00 = __builtin_amdgcn_wmma_f32_16x16x32_f16(
                        false, A0, false, B0, (short)0, acc00, false, false);
                    acc01 = __builtin_amdgcn_wmma_f32_16x16x32_f16(
                        false, A0, false, B1, (short)0, acc01, false, false);
                    acc10 = __builtin_amdgcn_wmma_f32_16x16x32_f16(
                        false, A1, false, B0, (short)0, acc10, false, false);
                    acc11 = __builtin_amdgcn_wmma_f32_16x16x32_f16(
                        false, A1, false, B1, (short)0, acc11, false, false);
                    a0 += FRAG; a1 += FRAG; bq0 += FRAG; bq1 += FRAG;
                }
                int m0 = ms * 32 + ((lane >> 4) << 3);
                int n0 = np * 32 + (lane & 15);
#pragma unroll
                for (int r = 0; r < 8; ++r) {
                    Y1[(long)(m0 + r) * NX + n0]           = acc00[r];
                    Y1[(long)(m0 + r) * NX + n0 + 16]      = acc01[r];
                    Y1[(long)(m0 + 16 + r) * NX + n0]      = acc10[r];
                    Y1[(long)(m0 + 16 + r) * NX + n0 + 16] = acc11[r];
                }
            } else {
                int j2 = job - jobs1;
                int ms = j2 / (NTH / 2), np = j2 % (NTH / 2);
                const _Float16* a0 = PF + (long)(ms * 2) * KTH * FRAG + lane * 16;
                const _Float16* a1 = PF + (long)(ms * 2 + 1) * KTH * FRAG + lane * 16;
                const _Float16* bq0 = WhF + (long)(np * 2) * KTH * FRAG + lane * 16;
                const _Float16* bq1 = WhF + (long)(np * 2 + 1) * KTH * FRAG + lane * 16;
                for (int kt = 0; kt < KTH; ++kt) {
                    __builtin_prefetch(bq0 + FRAG, 0, 3);
                    __builtin_prefetch(bq1 + FRAG, 0, 3);
                    v16h A0 = *(const v16h*)a0;
                    v16h A1 = *(const v16h*)a1;
                    v16h B0 = *(const v16h*)bq0;
                    v16h B1 = *(const v16h*)bq1;
                    acc00 = __builtin_amdgcn_wmma_f32_16x16x32_f16(
                        false, A0, false, B0, (short)0, acc00, false, false);
                    acc01 = __builtin_amdgcn_wmma_f32_16x16x32_f16(
                        false, A0, false, B1, (short)0, acc01, false, false);
                    acc10 = __builtin_amdgcn_wmma_f32_16x16x32_f16(
                        false, A1, false, B0, (short)0, acc10, false, false);
                    acc11 = __builtin_amdgcn_wmma_f32_16x16x32_f16(
                        false, A1, false, B1, (short)0, acc11, false, false);
                    a0 += FRAG; a1 += FRAG; bq0 += FRAG; bq1 += FRAG;
                }
                int m0 = ms * 32 + ((lane >> 4) << 3);
                int n0 = np * 32 + (lane & 15);
#pragma unroll
                for (int r = 0; r < 8; ++r) {
                    Y2[(long)(m0 + r) * NH + n0]           = acc00[r];
                    Y2[(long)(m0 + r) * NH + n0 + 16]      = acc01[r];
                    Y2[(long)(m0 + 16 + r) * NH + n0]      = acc10[r];
                    Y2[(long)(m0 + 16 + r) * NH + n0 + 16] = acc11[r];
                }
            }
        }
        target += gridDim.x; gsync(bar, target);

        // ---- Phase C: gates, h/c update, output head + argmax (block per node) ----
        for (int b = blockIdx.x; b < Bc; b += gridDim.x) {
            int t = order[s0 + b];
            int node = proc_order[t];
            int pid = parent[t];
            const float* y1 = Y1 + (long)b * NX;
            const float* y2 = Y2 + (long)b * NH;
            for (int j = tid; j < HID; j += blockDim.x) {
                float px = y1[j] + b_px[j];
                float ip = y1[HID + j]     + b_ioux[j]           + y2[j]           + b_iouh[j];
                float op = y1[2 * HID + j] + b_ioux[HID + j]     + y2[HID + j]     + b_iouh[HID + j];
                float up = y1[3 * HID + j] + b_ioux[2 * HID + j] + y2[2 * HID + j] + b_iouh[2 * HID + j];
                float fp = y1[4 * HID + j] + b_fx[j]             + y2[3 * HID + j] + b_fh[j];
                float iv = sigf(ip), ov = sigf(op), uv = tanhf(up), fv = sigf(fp);
                float pc = (pid < 0) ? 0.0f : cAll[(long)pid * HID + j];
                float c = iv * uv + fv * pc;
                float h = ov * tanhf(c) * sigf(px);
                cAll[(long)t * HID + j] = c;
                hAll[(long)t * HID + j] = (_Float16)h;
                hbuf[j] = h;
            }
            __syncthreads();
            for (int o = tid; o < NCLS; o += blockDim.x) {
                float d = b_out[o];
                const float* wr = Wout + (long)o * HID;
                for (int k = 0; k < HID; ++k) d += wr[k] * hbuf[k];
                dbuf[o] = d;
                outDist[(long)node * NCLS + o] = d;
            }
            __syncthreads();
            if (tid == 0) {
                int best = 1; float bv = dbuf[1];
                for (int o = 2; o < NCLS; ++o)
                    if (dbuf[o] > bv) { bv = dbuf[o]; best = o; }
                labels[t] = best;
                outLab[node] = (float)best;
            }
            __syncthreads();
        }
        target += gridDim.x; gsync(bar, target);
    }
}

extern "C" void kernel_launch(void* const* d_in, const int* in_sizes, int n_in,
                              void* d_out, int out_size, void* d_ws, size_t ws_size,
                              hipStream_t stream) {
    const float* features = (const float*)d_in[0];
    const float* embedW   = (const float*)d_in[1];
    const float* W_px     = (const float*)d_in[2];
    const float* b_px     = (const float*)d_in[3];
    const float* W_ioux   = (const float*)d_in[4];
    const float* b_ioux   = (const float*)d_in[5];
    const float* W_iouh   = (const float*)d_in[6];
    const float* b_iouh   = (const float*)d_in[7];
    const float* W_fx     = (const float*)d_in[8];
    const float* b_fx     = (const float*)d_in[9];
    const float* W_fh     = (const float*)d_in[10];
    const float* b_fh     = (const float*)d_in[11];
    const float* W_out    = (const float*)d_in[12];
    const float* b_out    = (const float*)d_in[13];
    const int* proc_order = (const int*)d_in[14];
    const int* parent     = (const int*)d_in[15];

    char* ws = (char*)d_ws;
    int*      ctl  = (int*)(ws + OFF_CTL);
    int*      chkp = (int*)(ws + OFF_CHK);
    int*      ord  = (int*)(ws + OFF_ORD);
    int*      dep  = (int*)(ws + OFF_DEP);
    int*      cnt  = (int*)(ws + OFF_CNT);
    int*      lab  = (int*)(ws + OFF_LAB);
    _Float16* WxF  = (_Float16*)(ws + OFF_WXF);
    _Float16* WhF  = (_Float16*)(ws + OFF_WHF);
    _Float16* hAll = (_Float16*)(ws + OFF_HAL);
    float*    cAll = (float*)(ws + OFF_CAL);
    _Float16* XF   = (_Float16*)(ws + OFF_XF);
    _Float16* PF   = (_Float16*)(ws + OFF_PF);
    float*    Y1   = (float*)(ws + OFF_Y1);
    float*    Y2   = (float*)(ws + OFF_Y2);

    float* outDist = (float*)d_out;
    float* outLab  = (float*)d_out + (long)NOBJ * NCLS;

    prep_weights<<<dim3(1024), dim3(256), 0, stream>>>(W_px, W_ioux, W_fx, W_iouh, W_fh,
                                                       WxF, WhF);
    prep_levels<<<dim3(1), dim3(64), 0, stream>>>(parent, ctl, chkp, ord, dep, cnt);
    tree_lstm_main<<<dim3(GBLK), dim3(TPB), 0, stream>>>(
        features, embedW, b_px, b_ioux, b_iouh, b_fx, b_fh, W_out, b_out,
        proc_order, parent, ctl, chkp, ord, lab, WxF, WhF, hAll, cAll,
        XF, PF, Y1, Y2, outDist, outLab, (unsigned*)ctl);
}
